// MoBA_4681514353439
// MI455X (gfx1250) — matrix operand order repaired and checked by reading, not compiled, not run on hardware
//
#include <hip/hip_runtime.h>
#include <hip/hip_bf16.h>

typedef __bf16 bf16;
typedef __attribute__((ext_vector_type(16))) __bf16 v16bf;
typedef __attribute__((ext_vector_type(8)))  __bf16 v8bf;
typedef __attribute__((ext_vector_type(8)))  float   v8f;

#define S_    2048
#define D_    2048
#define H_    16
#define DH    128
#define CHUNK_ 256
#define NBLK  8
#define TOPK_ 4
#define NEGF  (-1e30f)
#define APAD  40     // LDS row pitch (elems): 80B, 16B-aligned chunks, conflict-reducing

// Async global->LDS path (ASYNCcnt) if the toolchain exposes the builtin.
#if defined(__has_builtin)
#if __has_builtin(__builtin_amdgcn_global_load_async_to_lds_b128)
#define HAVE_ASYNC_LDS 1
#endif
#endif
#ifndef HAVE_ASYNC_LDS
#define HAVE_ASYNC_LDS 0
#endif

// Builtin wants int4 (gcc vector_size style) pointers: global src (as1), LDS dst (as3).
typedef int v4i_vs __attribute__((vector_size(16)));
typedef __attribute__((address_space(1))) v4i_vs as1_v4i;
typedef __attribute__((address_space(3))) v4i_vs as3_v4i;

// ---------------------------------------------------------------------------
// WMMA helpers (CDNA5 gfx1250, wave32)
// ---------------------------------------------------------------------------
static __device__ __forceinline__ v8f wmma_bf16(v16bf a, v16bf b, v8f c) {
  return __builtin_amdgcn_wmma_f32_16x16x32_bf16(false, a, false, b,
                                                 (short)0, c, false, false);
}

// A-frag (16x32 bf16) from a row-major, K-contiguous global matrix.
static __device__ __forceinline__ v16bf load_a_frag(const bf16* __restrict__ base,
                                                    int ld, int row, int k0, int lane) {
  const bf16* p = base + (size_t)(row + (lane & 15)) * ld + k0 + ((lane >> 4) << 3);
  v8bf lo = *(const v8bf*)p;
  v8bf hi = *(const v8bf*)(p + 16);
  v16bf r;
#pragma unroll
  for (int i = 0; i < 8; ++i) { r[i] = lo[i]; r[8 + i] = hi[i]; }
  return r;
}

// B-frag (32x16 bf16) from an NT source (K contiguous per column-row).
static __device__ __forceinline__ v16bf load_b_frag(const bf16* __restrict__ base,
                                                    int ld, int col, int k0, int lane) {
  const bf16* p = base + (size_t)(col + (lane & 15)) * ld + k0 + ((lane >> 4) << 3);
  v8bf lo = *(const v8bf*)p;
  v8bf hi = *(const v8bf*)(p + 16);
  v16bf r;
#pragma unroll
  for (int i = 0; i < 8; ++i) { r[i] = lo[i]; r[8 + i] = hi[i]; }
  return r;
}

// ---------------------------------------------------------------------------
// C = A(MxK bf16) * B(NxK bf16)^T, f32 accumulate.
// Block = 256 threads (8 waves); block tile 128x128; wave tile 32x64.
// A/B tiles double-buffered in LDS, staged with async global->LDS copies.
// ---------------------------------------------------------------------------
template <bool BF16OUT>
__global__ __launch_bounds__(256) void gemm_nt(const bf16* __restrict__ A,
                                               const bf16* __restrict__ B,
                                               void* __restrict__ Cout,
                                               int M, int N, int K) {
  __shared__ bf16 As[2][128][APAD];
  __shared__ bf16 Bs[2][128][APAD];

  const int lane = threadIdx.x & 31;
  const int wave = threadIdx.x >> 5;
  const int wm = wave >> 1;                 // 0..3
  const int wn = wave & 1;                  // 0..1
  const int mBase = blockIdx.y * 128;
  const int nBase = blockIdx.x * 128;
  const int m0l = wm * 32;                  // wave tile offsets within block tile
  const int n0l = wn * 64;

  // stage one 128x32 A tile + 128x32 B tile into LDS buffer `buf`
  auto stage = [&](int buf, int k0) {
#pragma unroll
    for (int it = 0; it < 2; ++it) {
      const int idx = (int)threadIdx.x + it * 256;   // 512 chunks of 8 elems
      const int row = idx >> 2;
      const int ch  = (idx & 3) << 3;
      const bf16* ga = A + (size_t)(mBase + row) * K + k0 + ch;
      const bf16* gb = B + (size_t)(nBase + row) * K + k0 + ch;
      bf16* la = &As[buf][row][ch];
      bf16* lb = &Bs[buf][row][ch];
#if HAVE_ASYNC_LDS
      __builtin_amdgcn_global_load_async_to_lds_b128((as1_v4i*)ga, (as3_v4i*)la, 0, 0);
      __builtin_amdgcn_global_load_async_to_lds_b128((as1_v4i*)gb, (as3_v4i*)lb, 0, 0);
#else
      *(v8bf*)la = *(const v8bf*)ga;
      *(v8bf*)lb = *(const v8bf*)gb;
#endif
    }
  };

  auto lda = [&](int buf, int rl) -> v16bf {
    const bf16* p = &As[buf][rl + (lane & 15)][(lane >> 4) << 3];
    v8bf lo = *(const v8bf*)p;
    v8bf hi = *(const v8bf*)(p + 16);
    v16bf r;
#pragma unroll
    for (int i = 0; i < 8; ++i) { r[i] = lo[i]; r[8 + i] = hi[i]; }
    return r;
  };
  auto ldb = [&](int buf, int cl) -> v16bf {
    const bf16* p = &Bs[buf][cl + (lane & 15)][(lane >> 4) << 3];
    v8bf lo = *(const v8bf*)p;
    v8bf hi = *(const v8bf*)(p + 16);
    v16bf r;
#pragma unroll
    for (int i = 0; i < 8; ++i) { r[i] = lo[i]; r[8 + i] = hi[i]; }
    return r;
  };

  v8f acc[2][4];
#pragma unroll
  for (int i = 0; i < 2; ++i)
#pragma unroll
    for (int j = 0; j < 4; ++j)
#pragma unroll
      for (int e = 0; e < 8; ++e) acc[i][j][e] = 0.0f;

  stage(0, 0);
  for (int k0 = 0; k0 < K; k0 += 32) {
    const int  buf  = (k0 >> 5) & 1;
    const bool more = (k0 + 32) < K;
    if (more) stage(buf ^ 1, k0 + 32);          // prefetch next tile (async)
#if HAVE_ASYNC_LDS
    if (more) asm volatile("s_wait_asynccnt 0x4" ::: "memory");  // current tile done
    else      asm volatile("s_wait_asynccnt 0x0" ::: "memory");
#else
    asm volatile("s_wait_dscnt 0x0" ::: "memory");
#endif
    __syncthreads();

    v16bf a0 = lda(buf, m0l);
    v16bf a1 = lda(buf, m0l + 16);
    v16bf b0 = ldb(buf, n0l);
    v16bf b1 = ldb(buf, n0l + 16);
    v16bf b2 = ldb(buf, n0l + 32);
    v16bf b3 = ldb(buf, n0l + 48);
    acc[0][0] = wmma_bf16(a0, b0, acc[0][0]);
    acc[0][1] = wmma_bf16(a0, b1, acc[0][1]);
    acc[0][2] = wmma_bf16(a0, b2, acc[0][2]);
    acc[0][3] = wmma_bf16(a0, b3, acc[0][3]);
    acc[1][0] = wmma_bf16(a1, b0, acc[1][0]);
    acc[1][1] = wmma_bf16(a1, b1, acc[1][1]);
    acc[1][2] = wmma_bf16(a1, b2, acc[1][2]);
    acc[1][3] = wmma_bf16(a1, b3, acc[1][3]);

    __syncthreads();                            // protect buf^1 before re-stage
  }

  const int hi = lane >> 4;
#pragma unroll
  for (int i = 0; i < 2; ++i)
#pragma unroll
    for (int j = 0; j < 4; ++j)
#pragma unroll
      for (int v = 0; v < 8; ++v) {
        const int row = mBase + m0l + i * 16 + v + hi * 8;
        const int col = nBase + n0l + j * 16 + (lane & 15);
        if (BF16OUT)
          ((bf16*)Cout)[(size_t)row * N + col] = (bf16)acc[i][j][v];
        else
          ((float*)Cout)[(size_t)row * N + col] = acc[i][j][v];
      }
}

// ---------------------------------------------------------------------------
// Elementwise f32 -> bf16 cast
// ---------------------------------------------------------------------------
__global__ void cast_f32_bf16(const float* __restrict__ src,
                              bf16* __restrict__ dst, int n) {
  int i = blockIdx.x * blockDim.x + threadIdx.x;
  if (i < n) dst[i] = (bf16)src[i];
}

// ---------------------------------------------------------------------------
// RoPE in-place on f32 q/k + bf16 copies for the WMMA kernels
// ---------------------------------------------------------------------------
__global__ void rope_kernel(float* __restrict__ qf, float* __restrict__ kf,
                            bf16* __restrict__ qb, bf16* __restrict__ kb) {
  int t = blockIdx.x * blockDim.x + threadIdx.x;
  const int total = S_ * H_ * (DH / 2);
  if (t >= total) return;
  const int ii = t & 63;
  const int h  = (t >> 6) & 15;
  const int s  = t >> 10;
  const float inv = __expf(-9.210340371976184f * (float)(2 * ii) * (1.0f / 128.0f));
  float sn, cs;
  __sincosf((float)s * inv, &sn, &cs);
  const size_t i1 = (size_t)s * D_ + (size_t)h * DH + ii;
  const size_t i2 = i1 + 64;
  const float q1 = qf[i1], q2 = qf[i2];
  const float k1 = kf[i1], k2 = kf[i2];
  const float qo1 = q1 * cs - q2 * sn, qo2 = q2 * cs + q1 * sn;
  const float ko1 = k1 * cs - k2 * sn, ko2 = k2 * cs + k1 * sn;
  qf[i1] = qo1; qf[i2] = qo2;
  kf[i1] = ko1; kf[i2] = ko2;
  qb[i1] = (bf16)qo1; qb[i2] = (bf16)qo2;
  kb[i1] = (bf16)ko1; kb[i2] = (bf16)ko2;
}

// ---------------------------------------------------------------------------
// Block means of roped k: km[blk][h][d]
// ---------------------------------------------------------------------------
__global__ void kmean_kernel(const float* __restrict__ kf, float* __restrict__ km) {
  const int blk = blockIdx.x >> 4;
  const int h   = blockIdx.x & 15;
  const int dcol = threadIdx.x;      // 128 threads
  float s = 0.0f;
  for (int t = 0; t < CHUNK_; ++t)
    s += kf[(size_t)(blk * CHUNK_ + t) * D_ + (size_t)h * DH + dcol];
  km[((size_t)blk * H_ + h) * DH + dcol] = s * (1.0f / CHUNK_);
}

// ---------------------------------------------------------------------------
// Gate + top-k block selection -> 8-bit mask per (h, s)
// ---------------------------------------------------------------------------
__global__ void gate_topk_kernel(const float* __restrict__ qf,
                                 const float* __restrict__ km,
                                 unsigned* __restrict__ selmask) {
  int t = blockIdx.x * blockDim.x + threadIdx.x;
  if (t >= H_ * S_) return;
  const int h = t / S_;
  const int s = t % S_;
  const float* qrow = qf + (size_t)s * D_ + (size_t)h * DH;
  const int qblk = s >> 8;
  float g[NBLK];
#pragma unroll
  for (int j = 0; j < NBLK; ++j) {
    if (j > qblk)  { g[j] = NEGF;  continue; }
    if (j == qblk) { g[j] = 1e30f; continue; }
    const float* kmr = km + ((size_t)j * H_ + h) * DH;
    float acc = 0.0f;
    for (int i = 0; i < DH; ++i) acc += qrow[i] * kmr[i];
    g[j] = acc;
  }
  unsigned msk = 0;
#pragma unroll
  for (int pick = 0; pick < TOPK_; ++pick) {
    int arg = 0; float best = -3.0e38f;
#pragma unroll
    for (int j = 0; j < NBLK; ++j)
      if (g[j] > best) { best = g[j]; arg = j; }
    if (best > NEGF * 0.5f) msk |= (1u << arg);
    g[arg] = -3.0e38f;
  }
  selmask[(size_t)h * S_ + s] = msk;
}

// ---------------------------------------------------------------------------
// v (S, H, d) f32 -> vT (H, d, S) bf16
// ---------------------------------------------------------------------------
__global__ void vtrans_kernel(const float* __restrict__ vf, bf16* __restrict__ vT) {
  int t = blockIdx.x * blockDim.x + threadIdx.x;
  if (t >= H_ * DH * S_) return;
  const int s    = t & (S_ - 1);
  const int dcol = (t >> 11) & (DH - 1);
  const int h    = t >> 18;
  vT[t] = (bf16)vf[(size_t)s * D_ + (size_t)h * DH + dcol];
}

// ---------------------------------------------------------------------------
// Sparse flash attention + fused RMS-norm
// ---------------------------------------------------------------------------
__global__ __launch_bounds__(256) void attn_kernel(const bf16* __restrict__ qb,
                                                   const bf16* __restrict__ kb,
                                                   const bf16* __restrict__ vT,
                                                   const unsigned* __restrict__ selmask,
                                                   const float* __restrict__ onw,
                                                   bf16* __restrict__ obn) {
  __shared__ bf16 pbuf[8][16][32];
  const int lane = threadIdx.x & 31;
  const int wave = threadIdx.x >> 5;
  const int hi   = lane >> 4;
  const int h    = blockIdx.y;
  const int q0   = blockIdx.x * 128 + wave * 16;

  unsigned rmask[8];
#pragma unroll
  for (int v = 0; v < 8; ++v)
    rmask[v] = selmask[(size_t)h * S_ + q0 + v + hi * 8];
  unsigned umask = 0;
#pragma unroll
  for (int v = 0; v < 8; ++v) umask |= rmask[v];
#pragma unroll
  for (int m = 1; m < 32; m <<= 1) umask |= __shfl_xor(umask, m, 32);

  v16bf qa[4];
#pragma unroll
  for (int kk = 0; kk < 4; ++kk)
    qa[kk] = load_a_frag(qb + (size_t)h * DH, D_, q0, kk * 32, lane);

  v8f oacc[8];
#pragma unroll
  for (int nf = 0; nf < 8; ++nf)
#pragma unroll
    for (int e = 0; e < 8; ++e) oacc[nf][e] = 0.0f;
  float mrow[8], lrow[8];
#pragma unroll
  for (int v = 0; v < 8; ++v) { mrow[v] = NEGF; lrow[v] = 0.0f; }

  const int qblk = q0 >> 8;
  const float scale = 0.08838834764831845f;   // 1/sqrt(128)

  for (int j = 0; j <= qblk; ++j) {
    if (!((umask >> j) & 1)) continue;
    const int kbase = j * CHUNK_;
    int kstop = kbase + CHUNK_;
    if (j == qblk) {
      int c = (q0 + 16 + 31) & ~31;
      if (c < kstop) kstop = c;
    }
    for (int kp = kbase; kp < kstop; kp += 32) {
      v8f sc0, sc1;
#pragma unroll
      for (int e = 0; e < 8; ++e) { sc0[e] = 0.0f; sc1[e] = 0.0f; }
#pragma unroll
      for (int kk = 0; kk < 4; ++kk) {
        v16bf bk0 = load_b_frag(kb + (size_t)h * DH, D_, kp,      kk * 32, lane);
        v16bf bk1 = load_b_frag(kb + (size_t)h * DH, D_, kp + 16, kk * 32, lane);
        sc0 = wmma_bf16(qa[kk], bk0, sc0);
        sc1 = wmma_bf16(qa[kk], bk1, sc1);
      }

      float corrv[8];
#pragma unroll
      for (int v = 0; v < 8; ++v) {
        const int row  = q0 + v + hi * 8;
        const int key0 = kp + (lane & 15);
        const bool sel = (rmask[v] >> j) & 1;
        float s0 = sc0[v] * scale;
        float s1 = sc1[v] * scale;
        if (!sel || key0      > row) s0 = NEGF;
        if (!sel || key0 + 16 > row) s1 = NEGF;
        float rmax = fmaxf(s0, s1);
#pragma unroll
        for (int m2 = 1; m2 < 16; m2 <<= 1)
          rmax = fmaxf(rmax, __shfl_xor(rmax, m2, 32));
        const float nm   = fmaxf(mrow[v], rmax);
        const float corr = __expf(mrow[v] - nm);
        const float p0 = (s0 <= NEGF * 0.5f) ? 0.0f : __expf(s0 - nm);
        const float p1 = (s1 <= NEGF * 0.5f) ? 0.0f : __expf(s1 - nm);
        float rs = p0 + p1;
#pragma unroll
        for (int m2 = 1; m2 < 16; m2 <<= 1) rs += __shfl_xor(rs, m2, 32);
        lrow[v] = lrow[v] * corr + rs;
        mrow[v] = nm;
        corrv[v] = corr;
        pbuf[wave][v + hi * 8][lane & 15]        = (bf16)p0;
        pbuf[wave][v + hi * 8][16 + (lane & 15)] = (bf16)p1;
      }
#pragma unroll
      for (int nf = 0; nf < 8; ++nf)
#pragma unroll
        for (int v = 0; v < 8; ++v) oacc[nf][v] *= corrv[v];

      asm volatile("s_wait_dscnt 0" ::: "memory");
      const bf16* prow = &pbuf[wave][lane & 15][0];
      v8bf plo = *(const v8bf*)(prow + hi * 16);
      v8bf phi = *(const v8bf*)(prow + hi * 16 + 8);
      v16bf pa;
#pragma unroll
      for (int e = 0; e < 8; ++e) { pa[e] = plo[e]; pa[8 + e] = phi[e]; }

      const bf16* vbase = vT + (size_t)h * DH * S_ + kp + hi * 16;
#pragma unroll
      for (int nf = 0; nf < 8; ++nf) {
        v16bf vb = *(const v16bf*)(vbase + (size_t)(nf * 16 + (lane & 15)) * S_);
        oacc[nf] = wmma_bf16(pa, vb, oacc[nf]);
      }
    }
  }

  float rnorm[8];
#pragma unroll
  for (int v = 0; v < 8; ++v) {
    const float invl = (lrow[v] > 0.0f) ? (1.0f / lrow[v]) : 0.0f;
#pragma unroll
    for (int nf = 0; nf < 8; ++nf) oacc[nf][v] *= invl;
    float ss = 0.0f;
#pragma unroll
    for (int nf = 0; nf < 8; ++nf) ss += oacc[nf][v] * oacc[nf][v];
#pragma unroll
    for (int m2 = 1; m2 < 16; m2 <<= 1) ss += __shfl_xor(ss, m2, 32);
    rnorm[v] = rsqrtf(ss * (1.0f / 128.0f) + 1e-6f);
  }
#pragma unroll
  for (int nf = 0; nf < 8; ++nf) {
    const int dcol = nf * 16 + (lane & 15);
    const float w = onw[dcol];
#pragma unroll
    for (int v = 0; v < 8; ++v) {
      const int row = q0 + v + hi * 8;
      obn[(size_t)row * D_ + (size_t)h * DH + dcol] =
          (bf16)(oacc[nf][v] * rnorm[v] * w);
    }
  }
}

// ---------------------------------------------------------------------------
// Host-side orchestration
// ---------------------------------------------------------------------------
extern "C" void kernel_launch(void* const* d_in, const int* in_sizes, int n_in,
                              void* d_out, int out_size, void* d_ws, size_t ws_size,
                              hipStream_t stream) {
  const float* X   = (const float*)d_in[0];
  const float* Wq  = (const float*)d_in[1];
  const float* Wk  = (const float*)d_in[2];
  const float* Wv  = (const float*)d_in[3];
  const float* Wo  = (const float*)d_in[4];
  const float* onw = (const float*)d_in[5];

  char* ws = (char*)d_ws;
  const size_t MB = (size_t)1 << 20;
  bf16*  Xb  = (bf16*)(ws + 0 * MB);
  bf16*  Wqb = (bf16*)(ws + 8 * MB);
  bf16*  Wkb = (bf16*)(ws + 16 * MB);
  bf16*  Wvb = (bf16*)(ws + 24 * MB);
  bf16*  Wob = (bf16*)(ws + 32 * MB);
  float* qf  = (float*)(ws + 40 * MB);
  float* kf  = (float*)(ws + 56 * MB);
  float* vf  = (float*)(ws + 72 * MB);
  bf16*  qbb = (bf16*)(ws + 88 * MB);
  bf16*  kbb = (bf16*)(ws + 96 * MB);
  bf16*  vT  = (bf16*)(ws + 104 * MB);
  bf16*  obn = (bf16*)(ws + 112 * MB);
  float* km  = (float*)(ws + 120 * MB);
  unsigned* selmask = (unsigned*)(ws + 120 * MB + 64 * 1024);

  const int nE = S_ * D_;
  cast_f32_bf16<<<(nE + 255) / 256, 256, 0, stream>>>(X,  Xb,  nE);
  cast_f32_bf16<<<(nE + 255) / 256, 256, 0, stream>>>(Wq, Wqb, nE);
  cast_f32_bf16<<<(nE + 255) / 256, 256, 0, stream>>>(Wk, Wkb, nE);
  cast_f32_bf16<<<(nE + 255) / 256, 256, 0, stream>>>(Wv, Wvb, nE);
  cast_f32_bf16<<<(nE + 255) / 256, 256, 0, stream>>>(Wo, Wob, nE);

  dim3 gg(D_ / 128, S_ / 128);
  gemm_nt<false><<<gg, 256, 0, stream>>>(Xb, Wqb, qf, S_, D_, D_);
  gemm_nt<false><<<gg, 256, 0, stream>>>(Xb, Wkb, kf, S_, D_, D_);
  gemm_nt<false><<<gg, 256, 0, stream>>>(Xb, Wvb, vf, S_, D_, D_);

  const int rth = S_ * H_ * (DH / 2);
  rope_kernel<<<(rth + 255) / 256, 256, 0, stream>>>(qf, kf, qbb, kbb);
  kmean_kernel<<<NBLK * H_, DH, 0, stream>>>(kf, km);
  gate_topk_kernel<<<(H_ * S_ + 255) / 256, 256, 0, stream>>>(qf, km, selmask);
  vtrans_kernel<<<(H_ * DH * S_ + 255) / 256, 256, 0, stream>>>(vf, vT);

  attn_kernel<<<dim3(S_ / 128, H_), 256, 0, stream>>>(qbb, kbb, vT, selmask, onw, obn);

  gemm_nt<false><<<gg, 256, 0, stream>>>(obn, Wob, (float*)d_out, S_, D_, D_);
}
